// SO3_Rotation_79439715106830
// MI455X (gfx1250) — matrix operand order repaired and checked by reading, not compiled, not run on hardware
//
#include <hip/hip_runtime.h>
#include <hip/hip_bf16.h>

typedef __attribute__((ext_vector_type(16))) _Float16 v16h;
typedef __attribute__((ext_vector_type(8)))  float    v8f;

#define LMAX  4
#define NROW  25      // (LMAX+1)^2
#define CCH   64
#define JTOT  165     // sum of (2l+1)^2

__constant__ int c_BOFF[6] = {0, 1, 10, 35, 84, 165};

// ---------------------------------------------------------------------------
// Setup kernel: compute J_l = expm( (pi/sqrt(2)) * (X0 + X1) ) per l, where
// X0/X1 are the real SO(3) generators obtained exactly as in the reference
// (su(2) generators conjugated by the real<->complex basis change Q).
// J is the Wigner-D of the pi-rotation about (x+y)/sqrt(2): J*J = I and
// J * exp(b*X1) * J = exp(b*X0).  165 floats total, written to workspace.
// Runs with 5 active threads; cost is negligible and fully deterministic.
// ---------------------------------------------------------------------------
__global__ void wigner_setup_J(float* __restrict__ Jout) {
  const int l = threadIdx.x;
  if (l > LMAX) return;
  const int n = 2 * l + 1;

  float Qr[81], Qi[81], Xr[81], Xi[81];
  for (int i = 0; i < n * n; ++i) { Qr[i] = Qi[i] = Xr[i] = Xi[i] = 0.f; }

  // Q = change_basis_real_to_complex(l)
  const float invs = 0.70710678118654752f;
  for (int m = -l; m < 0; ++m) {
    Qr[(l + m) * n + (l - m)] = invs;     // q[l+m, l+|m|] =  1/sqrt2
    Qi[(l + m) * n + (l + m)] = -invs;    // q[l+m, l-|m|] = -i/sqrt2
  }
  Qr[l * n + l] = 1.f;
  for (int m = 1; m <= l; ++m) {
    const float sg = (m & 1) ? -1.f : 1.f;
    Qr[(l + m) * n + (l + m)] = sg * invs;   // (-1)^m/sqrt2
    Qi[(l + m) * n + (l - m)] = sg * invs;   // i(-1)^m/sqrt2
  }
  // multiply by (-i)^l
  float fr, fi;
  switch (l & 3) { case 0: fr = 1;  fi = 0;  break;
                   case 1: fr = 0;  fi = -1; break;
                   case 2: fr = -1; fi = 0;  break;
                   default: fr = 0; fi = 1;  break; }
  for (int i = 0; i < n * n; ++i) {
    const float r = Qr[i], im = Qi[i];
    Qr[i] = fr * r - fi * im;
    Qi[i] = fr * im + fi * r;
  }

  // Xsum = Jx + Jy in the complex basis:
  //   Jx = 0.5(raising+lowering): antisym real tri-diagonal
  //   Jy = diag(i*m): imaginary diagonal
  for (int i = 0; i < n - 1; ++i) {
    const float mi = (float)(-l + i);
    const float s  = sqrtf((float)(l * (l + 1)) - mi * (mi + 1.f));
    Xr[i * n + (i + 1)]  =  0.5f * s;
    Xr[(i + 1) * n + i]  = -0.5f * s;
  }
  for (int i = 0; i < n; ++i) Xi[i * n + i] = (float)(-l + i);

  // T = Xsum * Q   (complex)
  float Tr[81], Ti[81];
  for (int i = 0; i < n; ++i)
    for (int j = 0; j < n; ++j) {
      float tr = 0.f, ti = 0.f;
      for (int k = 0; k < n; ++k) {
        const float xr = Xr[i * n + k], xi = Xi[i * n + k];
        const float qr = Qr[k * n + j], qi = Qi[k * n + j];
        tr += xr * qr - xi * qi;
        ti += xr * qi + xi * qr;
      }
      Tr[i * n + j] = tr; Ti[i * n + j] = ti;
    }

  // A = Re(Q^H T) * (pi/sqrt2) * 2^-6   (scaling for expm)
  float Amat[81];
  const float sc = 2.2214414690791831f * (1.0f / 64.0f);
  for (int i = 0; i < n; ++i)
    for (int j = 0; j < n; ++j) {
      float g = 0.f;
      for (int k = 0; k < n; ++k)
        g += Qr[k * n + i] * Tr[k * n + j] + Qi[k * n + i] * Ti[k * n + j];
      Amat[i * n + j] = g * sc;
    }

  // expm via Taylor(9) + 6 squarings  (||A|| <= 4*pi/64 ~ 0.2)
  float Eacc[81], Term[81], Tmp[81];
  for (int i = 0; i < n * n; ++i) { Eacc[i] = 0.f; Term[i] = 0.f; }
  for (int i = 0; i < n; ++i) { Eacc[i * n + i] = 1.f; Term[i * n + i] = 1.f; }
  for (int k = 1; k <= 9; ++k) {
    const float invk = 1.0f / (float)k;
    for (int i = 0; i < n; ++i)
      for (int j = 0; j < n; ++j) {
        float s2 = 0.f;
        for (int m = 0; m < n; ++m) s2 += Term[i * n + m] * Amat[m * n + j];
        Tmp[i * n + j] = s2 * invk;
      }
    for (int i = 0; i < n * n; ++i) { Term[i] = Tmp[i]; Eacc[i] += Tmp[i]; }
  }
  for (int sq = 0; sq < 6; ++sq) {
    for (int i = 0; i < n; ++i)
      for (int j = 0; j < n; ++j) {
        float s2 = 0.f;
        for (int m = 0; m < n; ++m) s2 += Eacc[i * n + m] * Eacc[m * n + j];
        Tmp[i * n + j] = s2;
      }
    for (int i = 0; i < n * n; ++i) Eacc[i] = Tmp[i];
  }
  for (int i = 0; i < n * n; ++i) Jout[c_BOFF[l] + i] = Eacc[i];
}

// ---------------------------------------------------------------------------
// Main kernel: 8 waves/block, one edge per wave.
//  1. per-lane (redundant) Euler extraction
//  2. wave-cooperative W = J * Yb(beta) * J into LDS (165 entries)
//  3. build f16 A fragments (Wigner D) directly in WMMA layout; stream
//     embedding into B fragments (branchless clamped loads); 2x4 tiles of
//     v_wmma_f32_16x16x32_f16; store f32 C with minimal predication.
// ---------------------------------------------------------------------------
__global__ __launch_bounds__(256)
void so3_rotate_wmma(const float* __restrict__ rot,
                     const float* __restrict__ emb,
                     const float* __restrict__ J,
                     float* __restrict__ out, int E) {
  __shared__ float Wlds[8][176];

  const int wave = threadIdx.x >> 5;
  const int lane = threadIdx.x & 31;
  const int r    = lane & 15;
  const int hi   = lane >> 4;
  const int e    = blockIdx.x * 8 + wave;
  const bool active = (e < E);

  __builtin_prefetch(J, 0, 3);   // warm L2 for the 660B J table

  float alpha = 0.f, beta = 0.f, gamma = 0.f;
  if (active) {
    const float* R = rot + (size_t)e * 9;
    // x = rot[:, :, 1]; normalize; clip; beta=acos(x1); alpha=atan2(x0, x2)
    float x0 = R[1], x1 = R[4], x2 = R[7];
    const float inrm = rsqrtf(x0 * x0 + x1 * x1 + x2 * x2);
    x0 = fminf(fmaxf(x0 * inrm, -1.f), 1.f);
    x1 = fminf(fmaxf(x1 * inrm, -1.f), 1.f);
    x2 = fminf(fmaxf(x2 * inrm, -1.f), 1.f);
    beta  = acosf(x1);
    alpha = atan2f(x0, x2);
    // gamma from row 0 of (rot_y(a)*rot_x(b))^T * R ; column 0 of Rab is (ca,0,-sa)
    float sa, ca; __sincosf(alpha, &sa, &ca);
    const float rp00 = ca * R[0] - sa * R[6];
    const float rp02 = ca * R[2] - sa * R[8];
    gamma = atan2f(rp02, rp00);

    // W = J * Yb * J, all 5 blocks packed (165 entries) into this wave's LDS
    for (int idx = lane; idx < JTOT; idx += 32) {
      int l = 0;
      while (idx >= c_BOFF[l + 1]) ++l;
      const int rel = idx - c_BOFF[l];
      const int n   = 2 * l + 1;
      const int p   = rel / n;
      const int q   = rel - p * n;
      const float* Jl = J + c_BOFF[l];
      float acc = 0.f;
      for (int k = 0; k < n; ++k) {
        float sbv, cbv;
        __sincosf((float)(l - k) * beta, &sbv, &cbv);
        acc += Jl[p * n + k] * (cbv * Jl[k * n + q] + sbv * Jl[(2 * l - k) * n + q]);
      }
      Wlds[wave][idx] = acc;
    }
  }
  __syncthreads();
  if (!active) return;

  // ---- A fragments (Wigner D as f16) in the ISA 16-bit A 16x32 layout ----
  // lane (r,hi) row M = 16*mt + r ; element t -> K = t + 8*((t>>3)+hi)
  v16h afrag[2];
  #pragma unroll
  for (int mt = 0; mt < 2; ++mt) {
    const int row = mt * 16 + r;
    int lr = -1, p = 0;
    if (row < NROW) {
      lr = 0;
      while (row >= (lr + 1) * (lr + 1)) ++lr;
      p = row - lr * lr;
    }
    float ca_p = 0.f, sa_p = 0.f;
    int n = 1, base = 0;
    if (lr >= 0) {
      __sincosf((float)(lr - p) * alpha, &sa_p, &ca_p);
      n = 2 * lr + 1;
      base = c_BOFF[lr];
    }
    const float* Wl = &Wlds[wave][base];
    #pragma unroll
    for (int t = 0; t < 16; ++t) {
      const int K = t + 8 * ((t >> 3) + hi);
      float v = 0.f;
      if (lr >= 0 && K >= lr * lr && K < (lr + 1) * (lr + 1)) {
        const int q = K - lr * lr;
        float sg_q, cg_q;
        __sincosf((float)(lr - q) * gamma, &sg_q, &cg_q);
        const float w_pq   = Wl[p * n + q];
        const float w_pqm  = Wl[p * n + (2 * lr - q)];
        const float w_mpq  = Wl[(2 * lr - p) * n + q];
        const float w_mpqm = Wl[(2 * lr - p) * n + (2 * lr - q)];
        v = ca_p * (cg_q * w_pq  - sg_q * w_pqm)
          + sa_p * (cg_q * w_mpq - sg_q * w_mpqm);
      }
      afrag[mt][t] = (_Float16)v;
    }
  }

  // ---- B fragments from embedding, WMMA, store C ----
  const size_t ebase = (size_t)e * NROW * CCH;
  #pragma unroll
  for (int nt = 0; nt < 4; ++nt) {
    const int col = nt * 16 + r;
    // B 32x16 f16: element t -> K = t + 8*((t>>3)+hi).
    // Branchless: load from clamped row, then select 0 for K >= 25 so the
    // loads form one unpredicated clause.
    v16h bfrag;
    #pragma unroll
    for (int t = 0; t < 16; ++t) {
      const int K  = t + 8 * ((t >> 3) + hi);
      const int Kc = (K < NROW) ? K : (NROW - 1);
      float v = emb[ebase + (size_t)Kc * CCH + col];
      v = (K < NROW) ? v : 0.f;
      bfrag[t] = (_Float16)v;
    }
    #pragma unroll
    for (int mt = 0; mt < 2; ++mt) {
      v8f c = {};
      c = __builtin_amdgcn_wmma_f32_16x16x32_f16(
              /*neg_a=*/false, afrag[mt], /*neg_b=*/false, bfrag,
              /*c_mod=*/(short)0, c, /*reuse_a=*/false, /*reuse_b=*/false);
      // C layout: elem k -> row M = mt*16 + k + 8*hi, col N = nt*16 + r.
      if (mt == 0) {
        // rows 0..15: always valid -> 8 unconditional stores
        #pragma unroll
        for (int k = 0; k < 8; ++k)
          out[ebase + (size_t)(k + 8 * hi) * CCH + col] = c[k];
      } else {
        // rows 16..31: k=0 gives row 16 (hi=0) or 24 (hi=1): always valid
        out[ebase + (size_t)(16 + 8 * hi) * CCH + col] = c[0];
        // k=1..7 valid only for hi==0 (rows 17..23): single divergence point
        if (hi == 0) {
          #pragma unroll
          for (int k = 1; k < 8; ++k)
            out[ebase + (size_t)(16 + k) * CCH + col] = c[k];
        }
      }
    }
  }
}

extern "C" void kernel_launch(void* const* d_in, const int* in_sizes, int n_in,
                              void* d_out, int out_size, void* d_ws, size_t ws_size,
                              hipStream_t stream) {
  (void)n_in; (void)out_size; (void)ws_size;
  const float* rot = (const float*)d_in[0];
  const float* emb = (const float*)d_in[1];
  float* out = (float*)d_out;
  float* J   = (float*)d_ws;          // 165 floats of per-l J matrices
  const int E = in_sizes[0] / 9;

  wigner_setup_J<<<1, 32, 0, stream>>>(J);
  const int blocks = (E + 7) / 8;     // 8 edges (waves) per 256-thread block
  so3_rotate_wmma<<<blocks, 256, 0, stream>>>(rot, emb, J, out, E);
}
